// DynamicTokenSideEmbedding_55602646614061
// MI455X (gfx1250) — matrix operand order repaired and checked by reading, not compiled, not run on hardware
//
#include <hip/hip_runtime.h>

#define SEQ     8192
#define CH      256
#define NCHUNK  32          // SEQ/CH
#define NSTAT   258         // 2 + 64+64 + 64+64
#define HDIM    64
#define DDIM    768
#define DHALF   384
#define INDIM   85
#define HSTRIDE 72          // padded f16 row stride (144B) -> conflict-free, 16B aligned

typedef _Float16 v8h  __attribute__((ext_vector_type(8)));
typedef _Float16 v16h __attribute__((ext_vector_type(16)));
typedef float    v8f  __attribute__((ext_vector_type(8)));

// ---------------- kernel 1: per-chunk partial histograms ----------------
__global__ __launch_bounds__(256) void k_partials(
    const int* __restrict__ token_ids, const int* __restrict__ var_ids,
    const float* __restrict__ t_has_int, const float* __restrict__ t_log,
    const int* __restrict__ v_fam, const int* __restrict__ v_gid,
    float* __restrict__ ws)
{
  __shared__ float part[NSTAT];
  const int t = threadIdx.x;
  for (int j = t; j < NSTAT; j += 256) part[j] = 0.f;
  __syncthreads();
  const int cg = blockIdx.x;           // row*32 + chunk
  const int p  = cg * CH + t;          // == row*SEQ + chunk*CH + t
  const int tok = token_ids[p];
  const int vid = var_ids[p];
  const float hi = t_has_int[tok];
  const bool valid = (hi > 0.f) && (tok != 0) && (tok != 1) && (tok != 2);
  const float vf  = valid ? 1.f : 0.f;
  const float tlv = t_log[tok] * vf;
  const int fam = v_fam[vid];
  const int gid = v_gid[vid];
  atomicAdd(&part[0], vf);
  atomicAdd(&part[1], tlv);
  atomicAdd(&part[2 + fam], vf);
  atomicAdd(&part[66 + fam], tlv);
  if (gid > 0) { atomicAdd(&part[130 + gid], vf); atomicAdd(&part[194 + gid], tlv); }
  __syncthreads();
  for (int j = t; j < NSTAT; j += 256) ws[cg * NSTAT + j] = part[j];
}

// ---------------- kernel 2: exclusive scan over chunks (per row) ----------------
__global__ __launch_bounds__(256) void k_scan(float* __restrict__ ws)
{
  const int row = blockIdx.x;
  for (int j = threadIdx.x; j < NSTAT; j += 256) {
    float run = 0.f;
    for (int c = 0; c < NCHUNK; ++c) {
      const int idx = (row * NCHUNK + c) * NSTAT + j;
      const float v = ws[idx];
      ws[idx] = run;
      run += v;
    }
  }
}

// ---------------- helper: accumulate one scaled W row into h ----------------
__device__ __forceinline__ void acc_row(float* __restrict__ h,
                                        const float* __restrict__ Wg1,
                                        int irow, float cf)
{
  const float* wr = Wg1 + irow * HDIM;
#pragma unroll
  for (int j = 0; j < HDIM; ++j) h[j] = __builtin_fmaf(cf, wr[j], h[j]);
}

// ---------------- kernel 3: fused features + LN + MLP (WMMA) ----------------
__global__ __launch_bounds__(256) void k_main(
    const int* __restrict__ token_ids, const int* __restrict__ var_ids,
    const float* __restrict__ t_has_int, const float* __restrict__ t_signed,
    const float* __restrict__ t_log,     const float* __restrict__ t_is_zero,
    const float* __restrict__ t_is_one,  const float* __restrict__ t_is_pow2,
    const int* __restrict__ v_fam,       const float* __restrict__ v_outer,
    const float* __restrict__ v_inner,   const float* __restrict__ v_has_o,
    const float* __restrict__ v_has_i,   const int* __restrict__ v_gid,
    const float* __restrict__ ln_g,      const float* __restrict__ ln_b,
    const float* __restrict__ W1,        const float* __restrict__ b1,
    const float* __restrict__ W2,        const float* __restrict__ b2,
    const float* __restrict__ scalep,
    const float* __restrict__ ws,        float* __restrict__ out)
{
  extern __shared__ char smem[];
  _Float16* W2T = (_Float16*)(smem);            // 384*72 halves  = 55296 B
  _Float16* hT  = (_Float16*)(smem + 55296);    // 256*72 halves  = 36864 B
  float* Wg1  = (float*)(smem + 92160);         // 85*64          = 21760 B
  float* Ccol = (float*)(smem + 113920);        // 64
  float* Gcol = (float*)(smem + 114176);        // 64
  float* b2s  = (float*)(smem + 114432);        // 768
  float* pfx  = (float*)(smem + 117504);        // 258
  float* sval = (float*)(smem + 118536);        // 256
  float* stlv = (float*)(smem + 119560);        // 256
  int*   sfam = (int*)  (smem + 120584);        // 256
  int*   sgid = (int*)  (smem + 121608);        // 256 -> total 122632 B

  const int t     = threadIdx.x;
  const int cg    = blockIdx.x;
  const int row   = cg >> 5;
  const int chunk = cg & 31;
  const int s     = chunk * CH + t;
  const int p     = row * SEQ + s;

  // --- cooperative preloads ---
  for (int idx = t; idx < INDIM * HDIM; idx += 256) {
    const int i = idx >> 6;
    Wg1[idx] = ln_g[i] * W1[idx];
  }
  if (t < HDIM) {
    float cc = b1[t], gg = 0.f;
    for (int i = 0; i < INDIM; ++i) {
      const float w = W1[i * HDIM + t];
      cc += ln_b[i] * w;
      gg += ln_g[i] * w;
    }
    Ccol[t] = cc; Gcol[t] = gg;
  }
  for (int idx = t; idx < DDIM; idx += 256) b2s[idx] = b2[idx];
  for (int idx = t; idx < NSTAT; idx += 256) pfx[idx] = ws[cg * NSTAT + idx];

  // --- per-position gathers ---
  const int tok = token_ids[p];
  const int vid = var_ids[p];
  const float has_int = t_has_int[tok];
  const float sgn     = t_signed[tok];
  const float tlog    = t_log[tok];
  const float is_zero = t_is_zero[tok];
  const float is_one  = t_is_one[tok];
  const float is_pow2 = t_is_pow2[tok];
  const int fam = v_fam[vid];
  const int gid = v_gid[vid];
  const float outer = v_outer[vid], inner = v_inner[vid];
  const float has_o = v_has_o[vid], has_i = v_has_i[vid];
  const bool valid = (has_int > 0.f) && (tok != 0) && (tok != 1) && (tok != 2);
  const float vf  = valid ? 1.f : 0.f;
  const float tlv = tlog * vf;
  float prev_tlog = 0.f, prev_sgn = 0.f;
  if (s > 0) { const int pt = token_ids[p - 1]; prev_tlog = t_log[pt]; prev_sgn = t_signed[pt]; }

  sval[t] = vf; stlv[t] = tlv; sfam[t] = fam; sgid[t] = gid;
  __syncthreads();

  // --- intra-chunk exclusive prefix (O(t) recompute) ---
  float cnt = 0.f, lsum = 0.f, fc = 0.f, fl = 0.f, gc = 0.f, gl = 0.f;
  for (int j = 0; j < t; ++j) {
    const float v = sval[j], lv = stlv[j];
    cnt += v; lsum += lv;
    if (sfam[j] == fam) { fc += v; fl += lv; }
    if (sgid[j] == gid) { gc += v; gl += lv; }
  }
  const float prev_count  = pfx[0] + cnt;
  const float prev_logsum = pfx[1] + lsum;
  const float psf_c = pfx[2 + fam] + fc;
  const float psf_l = pfx[66 + fam] + fl;
  float psg_c = 0.f, psg_l = 0.f;
  if (gid > 0) { psg_c = pfx[130 + gid] + gc; psg_l = pfx[194 + gid] + gl; }

  const float rd  = 1.0f / 8191.0f;
  const float pcm = fmaxf(prev_count, 1.0f);

  // 21 scalar features (+ implicit one-hot of value 1 at row 5+fam)
  float fv[21]; int fr[21];
  fr[0]=0;  fv[0]=has_int;        fr[1]=1;  fv[1]=sgn;
  fr[2]=2;  fv[2]=tlog;           fr[3]=3;  fv[3]=is_zero;
  fr[4]=4;  fv[4]=is_pow2;        fr[5]=69; fv[5]=outer;
  fr[6]=70; fv[6]=inner;          fr[7]=71; fv[7]=has_o;
  fr[8]=72; fv[8]=has_i;          fr[9]=73; fv[9]=(float)s * rd;
  fr[10]=74; fv[10]=prev_count * rd;   fr[11]=75; fv[11]=prev_logsum * rd;
  fr[12]=76; fv[12]=psf_c * rd;        fr[13]=77; fv[13]=psf_l * rd;
  fr[14]=78; fv[14]=psf_c / pcm;       fr[15]=79; fv[15]=psg_c * rd;
  fr[16]=80; fv[16]=psg_l * rd;        fr[17]=81; fv[17]=psg_c / pcm;
  fr[18]=82; fv[18]=prev_tlog;         fr[19]=83; fv[19]=prev_sgn;
  fr[20]=84; fv[20]=is_one;

  // LayerNorm moments (one-hot contributes exactly 1 to both sums)
  float sum = 1.0f, sq = 1.0f;
#pragma unroll
  for (int k = 0; k < 21; ++k) { sum += fv[k]; sq += fv[k] * fv[k]; }
  const float mu  = sum * (1.0f / INDIM);
  const float var = sq * (1.0f / INDIM) - mu * mu;
  const float inv = 1.0f / sqrtf(var + 1e-5f);

  // --- GEMM1 (22 rows), GELU, f16 stage into LDS ---
  float hreg[HDIM];
  const float minv = mu * inv;
#pragma unroll
  for (int j = 0; j < HDIM; ++j) hreg[j] = Ccol[j] - minv * Gcol[j];
#pragma unroll
  for (int k = 0; k < 21; ++k) acc_row(hreg, Wg1, fr[k], fv[k] * inv);
  acc_row(hreg, Wg1, 5 + fam, inv);

  _Float16* hrow = hT + t * HSTRIDE;
#pragma unroll
  for (int j = 0; j < HDIM; ++j) {
    float v = hreg[j];
    v = 0.5f * v * (1.0f + erff(v * 0.70710678118654752f));
    hrow[j] = (_Float16)v;
  }

  // --- GEMM2: (256x64) x (64x768) with v_wmma_f32_16x16x32_f16 ---
  const float sc = *scalep;
  const int w = t >> 5, lane = t & 31, lh = lane & 15, hv = lane >> 4;

  for (int half = 0; half < 2; ++half) {
    __syncthreads();                      // hT done / previous half's reads done
    for (int idx = t; idx < HDIM * DHALF; idx += 256) {
      const int k = idx / DHALF;
      const int nl = idx - k * DHALF;
      W2T[nl * HSTRIDE + k] = (_Float16)W2[k * DDIM + half * DHALF + nl];
    }
    __syncthreads();

#pragma unroll
    for (int mi2 = 0; mi2 < 2; ++mi2) {
      const int mi = 2 * w + mi2;
      // A fragments (16-bit A layout: halves = K[ks..ks+7] then K[ks+16..ks+23])
      const _Float16* ar = hT + (mi * 16 + lh) * HSTRIDE + hv * 8;
      v16h a0, a1;
      {
        const v8h l0 = *(const v8h*)(ar);
        const v8h h0 = *(const v8h*)(ar + 16);
        const v8h l1 = *(const v8h*)(ar + 32);
        const v8h h1 = *(const v8h*)(ar + 48);
#pragma unroll
        for (int i = 0; i < 8; ++i) {
          a0[i] = l0[i]; a0[i + 8] = h0[i];
          a1[i] = l1[i]; a1[i + 8] = h1[i];
        }
      }
      for (int nt = 0; nt < 24; ++nt) {
        // B fragments (same K layout as A: K[ks..ks+7] then K[ks+16..ks+23] per lane-half)
        const _Float16* br = W2T + (nt * 16 + lh) * HSTRIDE + hv * 8;
        const v8h b0l = *(const v8h*)(br);
        const v8h b0h = *(const v8h*)(br + 16);
        const v8h b1l = *(const v8h*)(br + 32);
        const v8h b1h = *(const v8h*)(br + 48);
        v16h bb0, bb1;
#pragma unroll
        for (int i = 0; i < 8; ++i) {
          bb0[i] = b0l[i]; bb0[i + 8] = b0h[i];
          bb1[i] = b1l[i]; bb1[i + 8] = b1h[i];
        }
        v8f c = {};
        c = __builtin_amdgcn_wmma_f32_16x16x32_f16(false, a0, false, bb0, (short)0, c, false, false);
        c = __builtin_amdgcn_wmma_f32_16x16x32_f16(false, a1, false, bb1, (short)0, c, false, false);

        const int col = half * DHALF + nt * 16 + lh;
        const float b2v = b2s[col];
        const int sM = chunk * CH + mi * 16 + hv * 8;     // local row in sequence
        const int base = row * SEQ + sM;
#pragma unroll
        for (int r = 0; r < 8; ++r) {
          out[(base + r) * DDIM + col] = sc * (c[r] + b2v);
        }
      }
    }
  }
}

// ---------------- launcher ----------------
extern "C" void kernel_launch(void* const* d_in, const int* in_sizes, int n_in,
                              void* d_out, int out_size, void* d_ws, size_t ws_size,
                              hipStream_t stream) {
  (void)in_sizes; (void)n_in; (void)out_size; (void)ws_size;
  const int*   token_ids = (const int*)  d_in[0];
  const int*   var_ids   = (const int*)  d_in[1];
  const float* t_has_int = (const float*)d_in[2];
  const float* t_signed  = (const float*)d_in[3];
  const float* t_log     = (const float*)d_in[4];
  const float* t_is_zero = (const float*)d_in[5];
  const float* t_is_one  = (const float*)d_in[6];
  const float* t_is_pow2 = (const float*)d_in[7];
  // d_in[8] = var_family_onehot : reconstructed from family id, unused
  const int*   v_fam     = (const int*)  d_in[9];
  const float* v_outer   = (const float*)d_in[10];
  const float* v_inner   = (const float*)d_in[11];
  const float* v_has_o   = (const float*)d_in[12];
  const float* v_has_i   = (const float*)d_in[13];
  const int*   v_gid     = (const int*)  d_in[14];
  const float* ln_g      = (const float*)d_in[15];
  const float* ln_b      = (const float*)d_in[16];
  const float* W1        = (const float*)d_in[17];
  const float* b1        = (const float*)d_in[18];
  const float* W2        = (const float*)d_in[19];
  const float* b2        = (const float*)d_in[20];
  const float* scalep    = (const float*)d_in[21];
  float* ws  = (float*)d_ws;
  float* out = (float*)d_out;

  k_partials<<<dim3(256), dim3(256), 0, stream>>>(token_ids, var_ids, t_has_int,
                                                  t_log, v_fam, v_gid, ws);
  k_scan<<<dim3(8), dim3(256), 0, stream>>>(ws);
  k_main<<<dim3(256), dim3(256), 122632, stream>>>(
      token_ids, var_ids, t_has_int, t_signed, t_log, t_is_zero, t_is_one,
      t_is_pow2, v_fam, v_outer, v_inner, v_has_o, v_has_i, v_gid,
      ln_g, ln_b, W1, b1, W2, b2, scalep, ws, out);
}
